// Attention_48747878809727
// MI455X (gfx1250) — matrix operand order repaired and checked by reading, not compiled, not run on hardware
//
#include <hip/hip_runtime.h>
#include <cstdint>
#include <cstddef>

#define NB_  32
#define NS_  2048
#define NE_  1024
#define ND_  512

typedef __bf16 v16bf __attribute__((ext_vector_type(16)));
typedef float  v8f   __attribute__((ext_vector_type(8)));

// round-to-nearest-even f32 -> bf16 (bit-exact fallback path)
__device__ __forceinline__ unsigned short f2bf(float f) {
  unsigned int u = __float_as_uint(f);
  u += 0x7FFFu + ((u >> 16) & 1u);
  return (unsigned short)(u >> 16);
}

// pack two f32 -> packed bf16x2; prefer native v_cvt_pk_bf16_f32
__device__ __forceinline__ unsigned pack_bf16(float lo, float hi) {
#if __has_builtin(__builtin_amdgcn_cvt_pk_bf16_f32)
  typedef __bf16 v2bf __attribute__((ext_vector_type(2)));
  union { v2bf v; unsigned u; } r;
  r.v = __builtin_amdgcn_cvt_pk_bf16_f32(lo, hi);
  return r.u;
#else
  return (unsigned)f2bf(lo) | ((unsigned)f2bf(hi) << 16);
#endif
}

// branch-free tanh; prefer CDNA5 native V_TANH_F32 trans op
__device__ __forceinline__ float fast_tanh(float x) {
#if __has_builtin(__builtin_amdgcn_tanhf)
  return __builtin_amdgcn_tanhf(x);
#elif __has_builtin(__builtin_amdgcn_tanh_f32)
  return __builtin_amdgcn_tanh_f32(x);
#else
  float t = __expf(2.0f * x);          // v_exp_f32; inf/0 saturate correctly
  return 1.0f - 2.0f / (t + 1.0f);
#endif
}

__device__ __forceinline__ v16bf make_frag(uint4 lo, uint4 hi) {
  union { uint4 q[2]; v16bf v; } u;
  u.q[0] = lo; u.q[1] = hi;
  return u.v;
}

// ---------------------------------------------------------------------------
// Prep 1: W_enc (f32, [1024][512] row-major) -> bf16 transposed [512 n][1024 k]
// ---------------------------------------------------------------------------
__global__ void prep_wenc(const float* __restrict__ Watt, unsigned short* __restrict__ WencT) {
  int idx = blockIdx.x * 256 + threadIdx.x;          // over 512*1024
  int n = idx >> 10;
  int k = idx & 1023;
  WencT[idx] = f2bf(Watt[k * ND_ + n]);
}

// ---------------------------------------------------------------------------
// Prep 2: dpb[b][d] = lh[b] . W_dec[:,d] + b_att[d]   (32x512, trivial)
// ---------------------------------------------------------------------------
__global__ void prep_dproj(const float* __restrict__ lh, const float* __restrict__ Watt,
                           const float* __restrict__ batt, float* __restrict__ dpb) {
  int d = blockIdx.x * 256 + threadIdx.x;            // 0..511 (grid.x == 2)
  int b = blockIdx.y;
  const float* Wd = Watt + (size_t)NE_ * ND_;        // W_att[1024:]
  float acc = batt[d];
  for (int k = 0; k < ND_; ++k)
    acc += lh[b * ND_ + k] * Wd[k * ND_ + d];
  dpb[b * ND_ + d] = acc;
}

// ---------------------------------------------------------------------------
// Main kernel: att[b][s] = tanh(enc[b][s]·W_enc + dpb[b]) · v  via bf16 WMMA
// grid (16, 32), 128 threads (4 waves), 256 KB dynamic LDS for the A panel.
// Each wave owns 32 rows; N processed in 128-wide chunks (acc[2][8] = 128 VGPRs).
// ---------------------------------------------------------------------------
__global__ void __launch_bounds__(128)
energy_kernel(const float* __restrict__ enc,
              const unsigned short* __restrict__ WencT,
              const float* __restrict__ dpb,
              const float* __restrict__ vvec,
              float* __restrict__ att) {
  extern __shared__ unsigned short Al[];             // [128 rows][1024 k] bf16
  const int b      = blockIdx.y;
  const int s_base = blockIdx.x * 128;
  const int tid    = threadIdx.x;
  const int wave   = tid >> 5;
  const int lane   = tid & 31;
  const int lane15 = lane & 15;
  const int kh     = lane >> 4;                      // which K-half this lane holds

  // Stage A panel: 128x1024 f32 -> bf16 in LDS (contiguous region of enc)
  const float4* src = (const float4*)(enc + ((size_t)b * NS_ + s_base) * NE_);
  uint2* dst = (uint2*)Al;
  for (int i = tid; i < 128 * NE_ / 4; i += 128) {
    float4 x = src[i];
    uint2 p;
    p.x = pack_bf16(x.x, x.y);
    p.y = pack_bf16(x.z, x.w);
    dst[i] = p;
  }
  __syncthreads();

  // A-fragment base pointers for this wave's two 16-row groups.
  // Layout (16-bit A 16x32): lanes0-15 hold K 0-7 & 16-23, lanes16-31 hold K 8-15 & 24-31.
  const unsigned short* a0base = Al + (size_t)(wave * 32 + lane15) * NE_;
  const unsigned short* a1base = a0base + 16 * NE_;

  float attacc[2][8];
  #pragma unroll
  for (int g = 0; g < 2; ++g)
    #pragma unroll
    for (int i = 0; i < 8; ++i) attacc[g][i] = 0.f;

  for (int chunk = 0; chunk < 4; ++chunk) {          // N in 128-wide chunks
    const int nb = chunk * 128;
    v8f acc[2][8];
    #pragma unroll
    for (int g = 0; g < 2; ++g)
      #pragma unroll
      for (int nt = 0; nt < 8; ++nt)
        acc[g][nt] = (v8f){0.f, 0.f, 0.f, 0.f, 0.f, 0.f, 0.f, 0.f};

    for (int kc = 0; kc < 32; ++kc) {                // K = 1024 in steps of 32
      const int ko = kc * 32;
      v16bf afr0 = make_frag(*(const uint4*)(a0base + ko + kh * 8),
                             *(const uint4*)(a0base + ko + 16 + kh * 8));
      v16bf afr1 = make_frag(*(const uint4*)(a1base + ko + kh * 8),
                             *(const uint4*)(a1base + ko + 16 + kh * 8));
      #pragma unroll
      for (int nt = 0; nt < 8; ++nt) {
        // B fragment: lane = column, K 8kh..8kh+7 and 16+8kh..16+8kh+7 (matches A)
        const unsigned short* bp =
            WencT + (size_t)(nb + nt * 16 + lane15) * NE_ + ko + kh * 8;
        v16bf bfr = make_frag(*(const uint4*)bp, *(const uint4*)(bp + 16));
        acc[0][nt] = __builtin_amdgcn_wmma_f32_16x16x32_bf16(
            false, afr0, false, bfr, (short)0, acc[0][nt], false, false);
        acc[1][nt] = __builtin_amdgcn_wmma_f32_16x16x32_bf16(
            false, afr1, false, bfr, (short)0, acc[1][nt], false, false);
      }
    }

    // Fused epilogue: tanh(acc + dec_proj + b_att) * v, accumulate over N
    #pragma unroll
    for (int nt = 0; nt < 8; ++nt) {
      const int col = nb + nt * 16 + lane15;
      const float dp = dpb[b * ND_ + col];
      const float vv = vvec[col];
      #pragma unroll
      for (int g = 0; g < 2; ++g)
        #pragma unroll
        for (int i = 0; i < 8; ++i) {
          float e = fast_tanh(acc[g][nt][i] + dp);
          attacc[g][i] += e * vv;
        }
    }
  }

  // Reduce partial dot over the 16 lanes of each half (C layout: lane = N col)
  #pragma unroll
  for (int g = 0; g < 2; ++g)
    #pragma unroll
    for (int i = 0; i < 8; ++i) {
      float x = attacc[g][i];
      x += __shfl_xor(x, 1);
      x += __shfl_xor(x, 2);
      x += __shfl_xor(x, 4);
      x += __shfl_xor(x, 8);
      attacc[g][i] = x;
    }
  // C layout: VGPR i -> M=i (lanes0-15) / M=8+i (lanes16-31)
  if (lane15 == 0) {
    #pragma unroll
    for (int g = 0; g < 2; ++g) {
      int r = s_base + wave * 32 + g * 16 + kh * 8;
      #pragma unroll
      for (int i = 0; i < 8; ++i)
        att[b * NS_ + r + i] = attacc[g][i];
    }
  }
}

// ---------------------------------------------------------------------------
// Softmax over S per batch (32 blocks x 256 threads, 8 elems/thread)
// ---------------------------------------------------------------------------
__global__ void softmax_kernel(const float* __restrict__ att, float* __restrict__ w) {
  int b = blockIdx.x, tid = threadIdx.x;
  __shared__ float red[8];
  const float* a = att + (size_t)b * NS_;
  float vals[8];
  float m = -1e30f;
  #pragma unroll
  for (int i = 0; i < 8; ++i) { vals[i] = a[tid + i * 256]; m = fmaxf(m, vals[i]); }
  for (int off = 16; off; off >>= 1) m = fmaxf(m, __shfl_xor(m, off));
  if ((tid & 31) == 0) red[tid >> 5] = m;
  __syncthreads();
  float bm = red[0];
  #pragma unroll
  for (int i = 1; i < 8; ++i) bm = fmaxf(bm, red[i]);
  __syncthreads();
  float s = 0.f;
  #pragma unroll
  for (int i = 0; i < 8; ++i) { vals[i] = __expf(vals[i] - bm); s += vals[i]; }
  for (int off = 16; off; off >>= 1) s += __shfl_xor(s, off);
  if ((tid & 31) == 0) red[tid >> 5] = s;
  __syncthreads();
  float bs = 0.f;
  #pragma unroll
  for (int i = 0; i < 8; ++i) bs += red[i];
  float inv = 1.0f / bs;
  #pragma unroll
  for (int i = 0; i < 8; ++i) w[(size_t)b * NS_ + tid + i * 256] = vals[i] * inv;
}

// ---------------------------------------------------------------------------
// context partials: split S into 8 chunks for bandwidth; deterministic (no atomics)
// part[sc][b][e] = sum over 256 s of w*enc.  grid (8, 32), 256 thr, float4/thread.
// ---------------------------------------------------------------------------
__global__ void ctx_partial(const float* __restrict__ enc, const float* __restrict__ w,
                            float* __restrict__ part) {
  int sc = blockIdx.x, b = blockIdx.y, tid = threadIdx.x;
  const float4* encb = (const float4*)(enc + ((size_t)b * NS_ + sc * 256) * NE_);
  const float*  wb   = w + (size_t)b * NS_ + sc * 256;
  float4 acc = {0.f, 0.f, 0.f, 0.f};
  for (int s = 0; s < 256; ++s) {
    float ws = wb[s];                 // uniform across block -> scalar load
    float4 x = encb[(size_t)s * 256 + tid];
    acc.x += ws * x.x; acc.y += ws * x.y; acc.z += ws * x.z; acc.w += ws * x.w;
  }
  ((float4*)part)[((size_t)sc * NB_ + b) * 256 + tid] = acc;
}

__global__ void ctx_reduce(const float* __restrict__ part, float* __restrict__ out) {
  int i = blockIdx.x * 256 + threadIdx.x;            // over 32*1024
  float s = 0.f;
  #pragma unroll
  for (int c = 0; c < 8; ++c) s += part[(size_t)c * NB_ * NE_ + i];
  out[i] = s;
}

// ---------------------------------------------------------------------------
extern "C" void kernel_launch(void* const* d_in, const int* in_sizes, int n_in,
                              void* d_out, int out_size, void* d_ws, size_t ws_size,
                              hipStream_t stream) {
  const float* enc  = (const float*)d_in[0];   // (32, 2048, 1024) f32
  const float* lh   = (const float*)d_in[1];   // (32, 512) f32
  const float* Watt = (const float*)d_in[2];   // (1536, 512) f32
  const float* batt = (const float*)d_in[3];   // (512,) f32
  const float* v    = (const float*)d_in[4];   // (512,) f32
  float* out = (float*)d_out;                  // (32, 1, 1024) f32

  char* ws = (char*)d_ws;
  unsigned short* WencT = (unsigned short*)(ws);             // 1 MB  bf16 [512][1024]
  float* dpb  = (float*)(ws + 1048576);                      // 64 KB
  float* att  = (float*)(ws + 1114112);                      // 256 KB
  float* w    = (float*)(ws + 1376256);                      // 256 KB
  float* part = (float*)(ws + 1638400);                      // 1 MB

  prep_wenc  <<<2048, 256, 0, stream>>>(Watt, WencT);
  prep_dproj <<<dim3(2, 32), 256, 0, stream>>>(lh, Watt, batt, dpb);
  energy_kernel<<<dim3(16, 32), 128, 128 * NE_ * sizeof(unsigned short), stream>>>(
      enc, WencT, dpb, v, att);
  softmax_kernel<<<NB_, 256, 0, stream>>>(att, w);
  ctx_partial<<<dim3(8, NB_), 256, 0, stream>>>(enc, w, part);
  ctx_reduce <<<128, 256, 0, stream>>>(part, out);
}
